// RecurrentGCN_21912923144886
// MI455X (gfx1250) — compile-verified, matrix operand order checked by reading, not run on hardware
//
#include <hip/hip_runtime.h>
#include <hip/hip_bf16.h>

typedef _Float16 v16h __attribute__((ext_vector_type(16)));
typedef float    v8f  __attribute__((ext_vector_type(8)));
typedef float    v4f  __attribute__((ext_vector_type(4)));

#define BLOCK_THREADS 256
#define WAVES_PER_BLOCK 8

// ---- LDS layout (units: halves) ----
// Wx fragments : 4 gates x 2 col-tiles x 4 k-steps x 512 halves
// Wh fragments : 4 gates x 2 col-tiles x 512
// Wlin fragment: 512 (cols 9..15 zero-padded)
// stage        : 8 waves x 512 (relu(h_new) in A-fragment layout)
#define WX_OFF    0
#define WH_OFF    (WX_OFF + 4*2*4*512)   // 16384
#define WLIN_OFF  (WH_OFF + 4*2*512)     // 20480
#define STAGE_OFF (WLIN_OFF + 512)       // 20992
#define LDS_HALVES (STAGE_OFF + 8*512)   // 25088 halves = 50176 B

// Native CDNA5 TANH transcendental when the builtin exists; exp fallback otherwise.
__device__ __forceinline__ float ftanh(float x) {
#if __has_builtin(__builtin_amdgcn_tanhf)
    return __builtin_amdgcn_tanhf(x);
#elif __has_builtin(__builtin_amdgcn_tanh_f32)
    return __builtin_amdgcn_tanh_f32(x);
#else
    float e = __expf(2.0f * x);
    return 1.0f - 2.0f / (e + 1.0f);
#endif
}
// sigmoid(x) = (1 + tanh(x/2)) / 2  -> one trans op
__device__ __forceinline__ float fsigmoid(float x) {
    return 0.5f * ftanh(0.5f * x) + 0.5f;
}

// k index owned by (lane L, half-slot p) in the 16-bit 16x32 WMMA operand layout:
//  L<16, p<8  -> k = p          ;  L<16, p>=8 -> k = p+8  (16..23)
//  L>=16,p<8  -> k = p+8 (8..15);  L>=16,p>=8 -> k = p+16 (24..31)
__device__ __forceinline__ int frag_k(int L, int p) {
    return p + 8 * ((p >= 8) + (L >= 16));
}

__global__ __launch_bounds__(BLOCK_THREADS)
void gconvlstm_wmma_kernel(
    const float* __restrict__ x,   const float* __restrict__ h,   const float* __restrict__ c,
    const float* __restrict__ Wx,  const float* __restrict__ bx,
    const float* __restrict__ Wh,  const float* __restrict__ bhp,
    const float* __restrict__ w_peep, const float* __restrict__ b_gate,
    const float* __restrict__ Wlin, const float* __restrict__ blin,
    float* __restrict__ out, float* __restrict__ hN, float* __restrict__ cN,
    int n)
{
    __shared__ __align__(64) _Float16 lds[LDS_HALVES];

    const int tid = threadIdx.x;

    // ---------- one-time per block: convert weights fp32->f16, pre-swizzled ----------
    // Wx [4,128,32] -> fragments (g, t, ks), element (k, n) = Wx[g][ks*32+k][t*16+n]
    for (int idx = tid; idx < 4*2*4*512; idx += BLOCK_THREADS) {
        int frag = idx >> 9, rem = idx & 511;
        int L = rem >> 4, p = rem & 15;
        int g = frag >> 3, t = (frag >> 2) & 1, ks = frag & 3;
        int nn = L & 15;
        int k = frag_k(L, p);
        lds[WX_OFF + idx] = (_Float16)Wx[((g * 128) + ks * 32 + k) * 32 + t * 16 + nn];
    }
    // Wh [4,32,32]
    for (int idx = tid; idx < 4*2*512; idx += BLOCK_THREADS) {
        int frag = idx >> 9, rem = idx & 511;
        int L = rem >> 4, p = rem & 15;
        int g = frag >> 1, t = frag & 1;
        int nn = L & 15;
        int k = frag_k(L, p);
        lds[WH_OFF + idx] = (_Float16)Wh[((g * 32) + k) * 32 + t * 16 + nn];
    }
    // Wlin [32,9] padded to 32x16
    for (int idx = tid; idx < 512; idx += BLOCK_THREADS) {
        int L = idx >> 4, p = idx & 15;
        int nn = L & 15;
        int k = frag_k(L, p);
        lds[WLIN_OFF + idx] = (_Float16)((nn < 9) ? Wlin[k * 9 + nn] : 0.0f);
    }
    __syncthreads();

    const int lane       = tid & 31;
    const int waveInBlk  = tid >> 5;
    const int hi         = lane >> 4;   // which K-half this lane owns
    const int nn         = lane & 15;   // row (A) / column (B,C,D) index
    const int waveGlob   = blockIdx.x * WAVES_PER_BLOCK + waveInBlk;
    const int waveStride = gridDim.x * WAVES_PER_BLOCK;
    const int numTiles   = (n + 15) >> 4;

    _Float16* stage = &lds[STAGE_OFF + waveInBlk * 512];

    // ---------- loop-invariant per-lane state (kept small: stay under 256 VGPRs) ----------
    // Only the Wlin B-fragment (8 VGPRs) is register-resident; Wh/Wx fragments are
    // re-read from LDS per tile so the wave stays out of the VGPR-MSB range.
    const v16h bwl = *(const v16h*)&lds[WLIN_OFF + lane * 16];

    // bias(g, col) = bx + bh + b_gate, per-lane constant per (g, col-tile)
    float biasv[4][2];
#pragma unroll
    for (int g = 0; g < 4; ++g)
#pragma unroll
        for (int t = 0; t < 2; ++t) {
            int col = t * 16 + nn;
            biasv[g][t] = bx[g * 32 + col] + bhp[g * 32 + col] + b_gate[g * 32 + col];
        }
    float wp[2][3];
#pragma unroll
    for (int t = 0; t < 2; ++t) {
        int col = t * 16 + nn;
        wp[t][0] = w_peep[col];
        wp[t][1] = w_peep[32 + col];
        wp[t][2] = w_peep[64 + col];
    }
    const float blv = (nn < 9) ? blin[nn] : 0.0f;

    for (int tile = waveGlob; tile < numTiles; tile += waveStride) {
        const int rowBase = tile << 4;
        const int myRow   = rowBase + nn;
        const int r       = (myRow < n) ? myRow : (n - 1);

        // prefetch next tile's slab of x (wave-uniform branch)
        int nextTile = tile + waveStride;
        if (nextTile < numTiles) {
            const float* pf = x + (size_t)(nextTile << 4) * 128 + lane * 64;
            __builtin_prefetch(pf, 0, 0);
            __builtin_prefetch(pf + 32, 0, 0);
        }

        // ---------- A fragments: x (4 k-steps of 32) and h (one k-step) ----------
        v16h ax[4];
#pragma unroll
        for (int ks = 0; ks < 4; ++ks) {
            const float* px = x + (size_t)r * 128 + ks * 32 + hi * 8;
            v4f f0 = *(const v4f*)(px);
            v4f f1 = *(const v4f*)(px + 4);
            v4f f2 = *(const v4f*)(px + 16);
            v4f f3 = *(const v4f*)(px + 20);
            v16h a;
#pragma unroll
            for (int q = 0; q < 4; ++q) {
                a[q]      = (_Float16)f0[q];
                a[4 + q]  = (_Float16)f1[q];
                a[8 + q]  = (_Float16)f2[q];
                a[12 + q] = (_Float16)f3[q];
            }
            ax[ks] = a;
        }
        const float* ph = h + (size_t)r * 32 + hi * 8;
        v4f h0 = *(const v4f*)(ph);
        v4f h1 = *(const v4f*)(ph + 4);
        v4f h2 = *(const v4f*)(ph + 16);
        v4f h3 = *(const v4f*)(ph + 20);
        v16h ah;
#pragma unroll
        for (int q = 0; q < 4; ++q) {
            ah[q]      = (_Float16)h0[q];
            ah[4 + q]  = (_Float16)h1[q];
            ah[8 + q]  = (_Float16)h2[q];
            ah[12 + q] = (_Float16)h3[q];
        }

        // ---------- gate pre-activations: 4 gates x 2 col-tiles ----------
        v8f acc[4][2];
#pragma unroll
        for (int g = 0; g < 4; ++g) {
#pragma unroll
            for (int t = 0; t < 2; ++t) {
                v8f a;
#pragma unroll
                for (int j = 0; j < 8; ++j) a[j] = biasv[g][t];
#pragma unroll
                for (int ks = 0; ks < 4; ++ks) {
                    v16h bw = *(const v16h*)&lds[WX_OFF + ((((g * 2 + t) * 4) + ks) << 9) + lane * 16];
                    a = __builtin_amdgcn_wmma_f32_16x16x32_f16(false, ax[ks], false, bw,
                                                               (short)0, a, false, false);
                }
                v16h bwh = *(const v16h*)&lds[WH_OFF + ((g * 2 + t) << 9) + lane * 16];
                a = __builtin_amdgcn_wmma_f32_16x16x32_f16(false, ah, false, bwh,
                                                           (short)0, a, false, false);
                acc[g][t] = a;
            }
        }

        // ---------- LSTM elementwise (f32), write h_new/c_new, stage relu(h_new) ----------
#pragma unroll
        for (int t = 0; t < 2; ++t) {
            int col = t * 16 + nn;
#pragma unroll
            for (int j = 0; j < 8; ++j) {
                int m   = j + hi * 8;
                int row = rowBase + m;
                float cv = c[(size_t)((row < n) ? row : (n - 1)) * 32 + col];
                float gi = fsigmoid(acc[0][t][j] + wp[t][0] * cv);
                float gf = fsigmoid(acc[1][t][j] + wp[t][1] * cv);
                float gt = ftanh(acc[2][t][j]);
                float cnew = gf * cv + gi * gt;
                float go = fsigmoid(acc[3][t][j] + wp[t][2] * cnew);
                float hnew = go * ftanh(cnew);
                if (row < n) {
                    cN[(size_t)row * 32 + col] = cnew;
                    hN[(size_t)row * 32 + col] = hnew;
                }
                float rh = hnew > 0.0f ? hnew : 0.0f;
                // scatter into A-fragment layout: row m, K index = col
                int sel   = (col >> 3) & 1;
                int lane2 = m + 16 * sel;
                int p2    = (col & 7) + ((col >= 16) ? 8 : 0);
                stage[lane2 * 16 + p2] = (_Float16)rh;
            }
        }
        // same-wave LDS RAW: wait for the scattered stores before the fragment read
        asm volatile("s_wait_dscnt 0" ::: "memory");

        // ---------- head: relu(h_new)[16x32] @ Wlin[32x16pad] ----------
        v16h ahd = *(const v16h*)&stage[lane * 16];
        v8f po;
#pragma unroll
        for (int j = 0; j < 8; ++j) po[j] = blv;
        po = __builtin_amdgcn_wmma_f32_16x16x32_f16(false, ahd, false, bwl,
                                                    (short)0, po, false, false);
        if (nn < 9) {
#pragma unroll
            for (int j = 0; j < 8; ++j) {
                int row = rowBase + j + hi * 8;
                if (row < n) out[(size_t)row * 9 + nn] = po[j];
            }
        }
    }
}

extern "C" void kernel_launch(void* const* d_in, const int* in_sizes, int n_in,
                              void* d_out, int out_size, void* d_ws, size_t ws_size,
                              hipStream_t stream) {
    const float* x      = (const float*)d_in[0];
    // d_in[1] = edge_index (int64), d_in[2] = edge_weight : dead for K=1 ChebConv
    const float* h      = (const float*)d_in[3];
    const float* c      = (const float*)d_in[4];
    const float* Wx     = (const float*)d_in[5];
    const float* bx     = (const float*)d_in[6];
    const float* Wh     = (const float*)d_in[7];
    const float* bhp    = (const float*)d_in[8];
    const float* w_peep = (const float*)d_in[9];
    const float* b_gate = (const float*)d_in[10];
    const float* Wlin   = (const float*)d_in[11];
    const float* blin   = (const float*)d_in[12];

    const int n = in_sizes[0] / 128;   // N from x's flat size

    float* out = (float*)d_out;              // [N,9]
    float* hN  = out + (size_t)n * 9;        // [N,32]
    float* cN  = hN + (size_t)n * 32;        // [N,32]

    const int blocks = 1024;                 // 8192 waves; grid-stride over 31250 tiles
    hipLaunchKernelGGL(gconvlstm_wmma_kernel, dim3(blocks), dim3(BLOCK_THREADS), 0, stream,
                       x, h, c, Wx, bx, Wh, bhp, w_peep, b_gate, Wlin, blin,
                       out, hN, cN, n);
}
